// Mamba2_16690242912541
// MI455X (gfx1250) — compile-verified, matrix operand order checked by reading, not run on hardware
//
#include <hip/hip_runtime.h>

#define DEV_INLINE __device__ __forceinline__

typedef __attribute__((ext_vector_type(16))) __bf16 v16bf;
typedef __attribute__((ext_vector_type(8)))  __bf16 v8bf;
typedef __attribute__((ext_vector_type(8)))  float  v8f;

constexpr int B_SZ = 2, SEQ = 4096, D_MODEL = 2048;
constexpr int D_INNER = 2048, D_XB = 512, D_CONV = 4;
constexpr int NHEADS = 16;
constexpr int CHUNK = 256, NC = SEQ / CHUNK;                 // 16 chunks
constexpr int CONV_DIM = D_INNER + 2 * D_XB;                 // 3072
constexpr int D_IN_PROJ = D_INNER + CONV_DIM + NHEADS;       // 5136
constexpr int NPAD1 = 5248;                                  // in_proj rows padded to 128-mult
constexpr int BL = B_SZ * SEQ;                               // 8192

DEV_INLINE v8f wmma_bf16(v16bf a, v16bf b, v8f c) {
  return __builtin_amdgcn_wmma_f32_16x16x32_bf16(false, a, false, b, (short)0, c, false, false);
}

// CDNA5 async DMA: copy 16 bytes global -> LDS, tracked by ASYNCcnt.
DEV_INLINE void async_cp16(const void* gptr, void* lptr) {
  unsigned lds = (unsigned)(uintptr_t)lptr;   // generic LDS addr: low 32 bits = LDS offset
  asm volatile("global_load_async_to_lds_b128 %0, %1, off"
               :: "v"(lds), "v"(gptr) : "memory");
}
DEV_INLINE void wait_async() {
  asm volatile("s_wait_asynccnt 0x0" ::: "memory");
}

// Load one 16x16x32 bf16 fragment row from LDS: two 16-byte runs.
DEV_INLINE v16bf ld_frag(const __bf16* __restrict__ row, int half8) {
  v8bf lo = *(const v8bf*)(row + half8);
  v8bf hi = *(const v8bf*)(row + half8 + 16);
  v16bf f;
#pragma unroll
  for (int i = 0; i < 8; ++i) { f[i] = lo[i]; f[i + 8] = hi[i]; }
  return f;
}

DEV_INLINE float siluf(float x) { return x / (1.f + __expf(-x)); }

// Inclusive cumsum of dt[l]*A over a 256-chunk into sAcs (Hillis-Steele).
DEV_INLINE void chunk_cumsum(float* sAcs, const float* __restrict__ dtsp,
                             int b, int c0, int h, float Aneg, int tid) {
  float dtv = dtsp[(size_t)(b * SEQ + c0 + tid) * NHEADS + h];
  sAcs[tid] = dtv * Aneg;
  __syncthreads();
  for (int off = 1; off < CHUNK; off <<= 1) {
    float v = (tid >= off) ? sAcs[tid - off] : 0.f;
    __syncthreads();
    sAcs[tid] += v;
    __syncthreads();
  }
}

// ---------------------------------------------------------------------------
// fp32 -> bf16 conversion helpers (run once per buffer).
// ---------------------------------------------------------------------------
__global__ __launch_bounds__(256) void to_bf16(
    const float* __restrict__ src, __bf16* __restrict__ dst) {
  size_t i = (size_t)blockIdx.x * 256 + threadIdx.x;
  dst[i] = (__bf16)src[i];
}

__global__ __launch_bounds__(256) void w_pad_bf16(
    const float* __restrict__ src, __bf16* __restrict__ dst, int rows, int cols) {
  size_t i = (size_t)blockIdx.x * 256 + threadIdx.x;  // over padrows*cols
  int r = (int)(i / cols);
  dst[i] = (r < rows) ? (__bf16)src[i] : (__bf16)0.f;
}

// ---------------------------------------------------------------------------
// Generic GEMM: C[M,N] = A[M,K] * W[N,K]^T  (bf16 in, WMMA, OutT out)
// Block tile 128x128, 8 waves (4 in M x 2 in N), K staged 64-wide via
// async-to-LDS DMA (no VGPR round trip, no converts).  W is pre-padded.
// ---------------------------------------------------------------------------
template <typename OutT>
__global__ __launch_bounds__(256) void gemm_bf16_nt(
    const __bf16* __restrict__ A, const __bf16* __restrict__ W,
    OutT* __restrict__ C, int M, int N, int K) {
  __shared__ __align__(16) __bf16 sA[128][72];
  __shared__ __align__(16) __bf16 sB[128][72];
  const int tid = threadIdx.x;
  const int lane = tid & 31, wave = tid >> 5;
  const int wm = wave & 3, wn = wave >> 2;
  const int mrow = lane & 15, half8 = (lane >> 4) * 8;
  const int blockM = blockIdx.x * 128, blockN = blockIdx.y * 128;
  v8f acc[2][4] = {};
  for (int k0 = 0; k0 < K; k0 += 64) {
#pragma unroll
    for (int i = 0; i < 4; ++i) {
      int idx = tid + i * 256;             // 0..1023 (128 rows x 8 16B-segs)
      int row = idx >> 3;
      int seg = (idx & 7) * 8;             // halves
      const __bf16* ga = A + (size_t)(blockM + row) * K + k0 + seg;
      const __bf16* gw = W + (size_t)(blockN + row) * K + k0 + seg;
      async_cp16(ga, &sA[row][seg]);
      async_cp16(gw, &sB[row][seg]);
      if (k0 + 64 < K) {                   // prefetch next stage
        __builtin_prefetch(ga + 64, 0, 1);
        __builtin_prefetch(gw + 64, 0, 1);
      }
    }
    wait_async();
    __syncthreads();
#pragma unroll
    for (int kk = 0; kk < 64; kk += 32) {
      v16bf bfr[4];
#pragma unroll
      for (int nt = 0; nt < 4; ++nt)
        bfr[nt] = ld_frag(&sB[wn * 64 + nt * 16 + mrow][kk], half8);
#pragma unroll
      for (int mt = 0; mt < 2; ++mt) {
        v16bf afr = ld_frag(&sA[wm * 32 + mt * 16 + mrow][kk], half8);
#pragma unroll
        for (int nt = 0; nt < 4; ++nt)
          acc[mt][nt] = wmma_bf16(afr, bfr[nt], acc[mt][nt]);
      }
    }
    __syncthreads();
  }
#pragma unroll
  for (int mt = 0; mt < 2; ++mt)
#pragma unroll
    for (int nt = 0; nt < 4; ++nt)
#pragma unroll
      for (int j = 0; j < 8; ++j) {
        int r = blockM + wm * 32 + mt * 16 + (lane >> 4) * 8 + j;
        int c = blockN + wn * 64 + nt * 16 + mrow;
        if (c < N) C[(size_t)r * N + c] = (OutT)acc[mt][nt][j];
      }
}

// ---------------------------------------------------------------------------
// Depthwise causal conv (width 4) + SiLU over the xBC slice of zxbcdt (bf16).
// ---------------------------------------------------------------------------
__global__ __launch_bounds__(256) void conv_silu(
    const __bf16* __restrict__ zx, const float* __restrict__ cw,
    const float* __restrict__ cb, __bf16* __restrict__ xbc) {
  size_t idx = (size_t)blockIdx.x * 256 + threadIdx.x;   // < BL*CONV_DIM
  int ch = (int)(idx % CONV_DIM);
  int bl = (int)(idx / CONV_DIM);
  int t = bl & (SEQ - 1);
  float acc = cb[ch];
#pragma unroll
  for (int k = 0; k < D_CONV; ++k) {
    int tt = t - (D_CONV - 1) + k;
    if (tt >= 0)
      acc += (float)zx[(size_t)(bl - (D_CONV - 1) + k) * D_IN_PROJ + D_INNER + ch] *
             cw[ch * D_CONV + k];
  }
  xbc[idx] = (__bf16)siluf(acc);
}

__global__ __launch_bounds__(256) void dt_softplus(
    const __bf16* __restrict__ zx, const float* __restrict__ dt_bias,
    float* __restrict__ dtsp) {
  int idx = blockIdx.x * 256 + threadIdx.x;              // < BL*NHEADS
  int hh = idx & 15;
  int bl = idx >> 4;
  float v = (float)zx[(size_t)bl * D_IN_PROJ + D_INNER + CONV_DIM + hh] + dt_bias[hh];
  dtsp[idx] = (v > 20.f) ? v : log1pf(__expf(v));
}

// ---------------------------------------------------------------------------
// Y_diag: per (b, chunk, head, 64-row block).
// G = C*B^T (WMMA), masked decay, then Y = G*xdt (WMMA).
// C and B tiles DMA'd straight to LDS (bf16 global), xdt scaled manually.
// ---------------------------------------------------------------------------
__global__ __launch_bounds__(256) void ssd_diag(
    const __bf16* __restrict__ xBC, const float* __restrict__ dtsp,
    const float* __restrict__ A_log, float* __restrict__ ybuf) {
  __shared__ __align__(16) float  sAcs[CHUNK];
  __shared__ __align__(16) __bf16 sC[64][136];
  __shared__ __align__(16) __bf16 sB[64][136];
  __shared__ __align__(16) __bf16 sXdtT[128][72];   // [p][l] transposed
  __shared__ __align__(16) __bf16 sG[64][72];
  const int bid = blockIdx.x;
  const int rb = bid & 3, h = (bid >> 2) & 15, c = (bid >> 6) & 15, b = bid >> 10;
  const int c0 = c * CHUNK, kv = h >> 2;
  const int tid = threadIdx.x;
  const int lane = tid & 31, wave = tid >> 5;
  const int wm = wave & 1, wn = wave >> 1;
  const int mrow = lane & 15, half8 = (lane >> 4) * 8;
  const float Aneg = -__expf(A_log[h]);

#pragma unroll
  for (int i = 0; i < 4; ++i) {                      // C rows: async DMA
    int idx = tid + i * 256, row = idx >> 4, seg = (idx & 15) * 8;
    int l = c0 + rb * 64 + row;
    async_cp16(xBC + (size_t)(b * SEQ + l) * CONV_DIM + 2 * D_XB + h * 128 + seg,
               &sC[row][seg]);
  }
  chunk_cumsum(sAcs, dtsp, b, c0, h, Aneg, tid);

  v8f yacc[2][2] = {};
  for (int st = 0; st <= rb; ++st) {
    __syncthreads();
#pragma unroll
    for (int i = 0; i < 4; ++i) {                    // B tile: async DMA
      int idx = tid + i * 256, row = idx >> 4, seg = (idx & 15) * 8;
      int l = c0 + st * 64 + row;
      async_cp16(xBC + (size_t)(b * SEQ + l) * CONV_DIM + D_XB + kv * 128 + seg,
                 &sB[row][seg]);
    }
#pragma unroll
    for (int i = 0; i < 4; ++i) {                    // xdt^T tile: scale+transpose
      int idx = tid + i * 256, row = idx >> 4, c8 = (idx & 15) * 8;
      int l = c0 + st * 64 + row;
      float dtv = dtsp[(size_t)(b * SEQ + l) * NHEADS + h];
      v8bf vx = *(const v8bf*)(xBC + (size_t)(b * SEQ + l) * CONV_DIM + kv * 128 + c8);
#pragma unroll
      for (int j = 0; j < 8; ++j)
        sXdtT[c8 + j][row] = (__bf16)((float)vx[j] * dtv);
    }
    wait_async();
    __syncthreads();
    v8f gacc[2] = {};                                // G = C*B^T (64x64)
#pragma unroll
    for (int kk = 0; kk < 128; kk += 32) {
      v16bf bfr = ld_frag(&sB[wn * 16 + mrow][kk], half8);
#pragma unroll
      for (int mt = 0; mt < 2; ++mt) {
        v16bf afr = ld_frag(&sC[wm * 32 + mt * 16 + mrow][kk], half8);
        gacc[mt] = wmma_bf16(afr, bfr, gacc[mt]);
      }
    }
#pragma unroll
    for (int mt = 0; mt < 2; ++mt)
#pragma unroll
      for (int j = 0; j < 8; ++j) {
        int rl = wm * 32 + mt * 16 + (lane >> 4) * 8 + j;
        int sl = wn * 16 + mrow;
        int r = rb * 64 + rl, s = st * 64 + sl;
        float g = (s <= r) ? gacc[mt][j] * __expf(sAcs[r] - sAcs[s]) : 0.f;
        sG[rl][sl] = (__bf16)g;
      }
    __syncthreads();
#pragma unroll
    for (int kk = 0; kk < 64; kk += 32) {            // Y += G*xdt
      v16bf afr[2], bfr[2];
#pragma unroll
      for (int mt = 0; mt < 2; ++mt)
        afr[mt] = ld_frag(&sG[wm * 32 + mt * 16 + mrow][kk], half8);
#pragma unroll
      for (int nt = 0; nt < 2; ++nt)
        bfr[nt] = ld_frag(&sXdtT[wn * 32 + nt * 16 + mrow][kk], half8);
#pragma unroll
      for (int mt = 0; mt < 2; ++mt)
#pragma unroll
        for (int nt = 0; nt < 2; ++nt)
          yacc[mt][nt] = wmma_bf16(afr[mt], bfr[nt], yacc[mt][nt]);
    }
  }
#pragma unroll
  for (int mt = 0; mt < 2; ++mt)
#pragma unroll
    for (int nt = 0; nt < 2; ++nt)
#pragma unroll
      for (int j = 0; j < 8; ++j) {
        int rl = wm * 32 + mt * 16 + (lane >> 4) * 8 + j;
        int p = wn * 32 + nt * 16 + mrow;
        int l = c0 + rb * 64 + rl;
        ybuf[((size_t)(b * SEQ + l) * NHEADS + h) * 128 + p] = yacc[mt][nt][j];
      }
}

// ---------------------------------------------------------------------------
// Per-chunk states[p][n] = sum_l xdt[l][p]*decay[l]*B[l][n]   (128x128, K=256)
// Both operands need transpose/scale -> manual LDS staging.
// ---------------------------------------------------------------------------
__global__ __launch_bounds__(256) void ssd_states(
    const __bf16* __restrict__ xBC, const float* __restrict__ dtsp,
    const float* __restrict__ A_log, float* __restrict__ states,
    float* __restrict__ cdec) {
  __shared__ __align__(16) float  sAcs[CHUNK];
  __shared__ __align__(16) __bf16 sXdtT[128][72];   // [p][l]
  __shared__ __align__(16) __bf16 sBT[128][72];     // [n][l]
  const int bid = blockIdx.x;
  const int h = bid & 15, c = (bid >> 4) & 15, b = bid >> 8;
  const int c0 = c * CHUNK, kv = h >> 2;
  const int tid = threadIdx.x;
  const int lane = tid & 31, wave = tid >> 5;
  const int wm = wave & 1, wn = wave >> 1;
  const int mrow = lane & 15, half8 = (lane >> 4) * 8;
  const float Aneg = -__expf(A_log[h]);

  chunk_cumsum(sAcs, dtsp, b, c0, h, Aneg, tid);
  const float alast = sAcs[CHUNK - 1];
  if (tid == 0) cdec[(b * NHEADS + h) * NC + c] = __expf(alast);

  v8f acc[4][2] = {};
  for (int st = 0; st < 4; ++st) {
    __syncthreads();
#pragma unroll
    for (int i = 0; i < 4; ++i) {
      int idx = tid + i * 256, row = idx >> 4, c8 = (idx & 15) * 8;
      int l = c0 + st * 64 + row;
      size_t base = (size_t)(b * SEQ + l) * CONV_DIM;
      float f = dtsp[(size_t)(b * SEQ + l) * NHEADS + h] *
                __expf(alast - sAcs[st * 64 + row]);
      v8bf vx = *(const v8bf*)(xBC + base + kv * 128 + c8);
      v8bf vb = *(const v8bf*)(xBC + base + D_XB + kv * 128 + c8);
#pragma unroll
      for (int j = 0; j < 8; ++j) {
        sXdtT[c8 + j][row] = (__bf16)((float)vx[j] * f);
        sBT[c8 + j][row] = vb[j];
      }
    }
    __syncthreads();
#pragma unroll
    for (int kk = 0; kk < 64; kk += 32) {
      v16bf bfr[2];
#pragma unroll
      for (int nt = 0; nt < 2; ++nt)
        bfr[nt] = ld_frag(&sBT[wn * 32 + nt * 16 + mrow][kk], half8);
#pragma unroll
      for (int mt = 0; mt < 4; ++mt) {
        v16bf afr = ld_frag(&sXdtT[wm * 64 + mt * 16 + mrow][kk], half8);
#pragma unroll
        for (int nt = 0; nt < 2; ++nt)
          acc[mt][nt] = wmma_bf16(afr, bfr[nt], acc[mt][nt]);
      }
    }
  }
  size_t base = (size_t)((b * NC + c) * NHEADS + h) * (128 * 128);
#pragma unroll
  for (int mt = 0; mt < 4; ++mt)
#pragma unroll
    for (int nt = 0; nt < 2; ++nt)
#pragma unroll
      for (int j = 0; j < 8; ++j) {
        int p = wm * 64 + mt * 16 + (lane >> 4) * 8 + j;
        int n = wn * 32 + nt * 16 + mrow;
        states[base + (size_t)p * 128 + n] = acc[mt][nt][j];
      }
}

// Sequential inter-chunk recurrence: prev[c] = carry; carry = carry*d[c]+s[c].
__global__ __launch_bounds__(256) void ssd_scan(
    const float* __restrict__ states, const float* __restrict__ cdec,
    float* __restrict__ prevb) {
  const int h = blockIdx.x & 15, b = blockIdx.x >> 4;
  const int tid = threadIdx.x;
  for (int e = 0; e < 64; ++e) {
    int idx = tid + e * 256;                         // 0..16383 in (p,n)
    float carry = 0.f;
    for (int c = 0; c < NC; ++c) {
      size_t o = (size_t)((b * NC + c) * NHEADS + h) * (128 * 128) + idx;
      prevb[o] = carry;
      carry = carry * cdec[(b * NHEADS + h) * NC + c] + states[o];
    }
  }
}

// ---------------------------------------------------------------------------
// Y_off = exp(Acs[l]) * (C*prev^T)  + D[h]*x, accumulated into ybuf.
// ---------------------------------------------------------------------------
__global__ __launch_bounds__(256) void ssd_off(
    const __bf16* __restrict__ xBC, const float* __restrict__ dtsp,
    const float* __restrict__ A_log, const float* __restrict__ prevb,
    const float* __restrict__ Dvec, float* __restrict__ ybuf) {
  __shared__ __align__(16) float  sAcs[CHUNK];
  __shared__ __align__(16) __bf16 sC[64][136];
  __shared__ __align__(16) __bf16 sP[128][136];     // prev [p][n]
  const int bid = blockIdx.x;
  const int rb = bid & 3, h = (bid >> 2) & 15, c = (bid >> 6) & 15, b = bid >> 10;
  const int c0 = c * CHUNK, kv = h >> 2;
  const int tid = threadIdx.x;
  const int lane = tid & 31, wave = tid >> 5;
  const int wm = wave & 1, wn = wave >> 1;
  const int mrow = lane & 15, half8 = (lane >> 4) * 8;
  const float Aneg = -__expf(A_log[h]);
  const float Dh = Dvec[h];

#pragma unroll
  for (int i = 0; i < 4; ++i) {                      // C rows: async DMA
    int idx = tid + i * 256, row = idx >> 4, seg = (idx & 15) * 8;
    int l = c0 + rb * 64 + row;
    async_cp16(xBC + (size_t)(b * SEQ + l) * CONV_DIM + 2 * D_XB + h * 128 + seg,
               &sC[row][seg]);
  }
  chunk_cumsum(sAcs, dtsp, b, c0, h, Aneg, tid);

  size_t pbase = (size_t)((b * NC + c) * NHEADS + h) * (128 * 128);
#pragma unroll
  for (int i = 0; i < 16; ++i) {                     // prev: fp32 -> bf16
    int idx = tid + i * 256, row = idx >> 5, c4 = (idx & 31) * 4;
    float4 v = *(const float4*)(prevb + pbase + (size_t)row * 128 + c4);
    sP[row][c4 + 0] = (__bf16)v.x; sP[row][c4 + 1] = (__bf16)v.y;
    sP[row][c4 + 2] = (__bf16)v.z; sP[row][c4 + 3] = (__bf16)v.w;
  }
  wait_async();
  __syncthreads();

  v8f acc[2][2] = {};
#pragma unroll
  for (int kk = 0; kk < 128; kk += 32) {
    v16bf bfr[2];
#pragma unroll
    for (int nt = 0; nt < 2; ++nt)
      bfr[nt] = ld_frag(&sP[wn * 32 + nt * 16 + mrow][kk], half8);
#pragma unroll
    for (int mt = 0; mt < 2; ++mt) {
      v16bf afr = ld_frag(&sC[wm * 32 + mt * 16 + mrow][kk], half8);
#pragma unroll
      for (int nt = 0; nt < 2; ++nt)
        acc[mt][nt] = wmma_bf16(afr, bfr[nt], acc[mt][nt]);
    }
  }
#pragma unroll
  for (int mt = 0; mt < 2; ++mt)
#pragma unroll
    for (int nt = 0; nt < 2; ++nt)
#pragma unroll
      for (int j = 0; j < 8; ++j) {
        int rl = wm * 32 + mt * 16 + (lane >> 4) * 8 + j;
        int p = wn * 32 + nt * 16 + mrow;
        int l = c0 + rb * 64 + rl;
        float ex = __expf(sAcs[rb * 64 + rl]);
        float xv = (float)xBC[(size_t)(b * SEQ + l) * CONV_DIM + kv * 128 + p];
        size_t yi = ((size_t)(b * SEQ + l) * NHEADS + h) * 128 + p;
        ybuf[yi] += acc[mt][nt][j] * ex + Dh * xv;
      }
}

// Gated per-head RMSNorm: yn = normw * (y*silu(z)) * rsqrt(mean_head(g^2)+eps)
__global__ __launch_bounds__(256) void gatenorm(
    const float* __restrict__ ybuf, const __bf16* __restrict__ zx,
    const float* __restrict__ normw, __bf16* __restrict__ ynb) {
  __shared__ float part[256];
  __shared__ float srstd[16];
  const int bl = blockIdx.x, tid = threadIdx.x;
  const float* yrow = ybuf + (size_t)bl * D_INNER;
  const __bf16* zrow = zx + (size_t)bl * D_IN_PROJ;
  float g[8], ss = 0.f;
#pragma unroll
  for (int i = 0; i < 8; ++i) {
    int d = tid * 8 + i;
    float gv = yrow[d] * siluf((float)zrow[d]);
    g[i] = gv; ss += gv * gv;
  }
  part[tid] = ss;
  __syncthreads();
  if (tid < 16) {
    float s = 0.f;
#pragma unroll
    for (int i = 0; i < 16; ++i) s += part[tid * 16 + i];
    srstd[tid] = rsqrtf(s / 128.f + 1e-5f);
  }
  __syncthreads();
  float r = srstd[tid >> 4];
#pragma unroll
  for (int i = 0; i < 8; ++i) {
    int d = tid * 8 + i;
    ynb[(size_t)bl * D_INNER + d] = (__bf16)(g[i] * r * normw[d]);
  }
}

extern "C" void kernel_launch(void* const* d_in, const int* in_sizes, int n_in,
                              void* d_out, int out_size, void* d_ws, size_t ws_size,
                              hipStream_t stream) {
  const float* hs        = (const float*)d_in[0];
  const float* in_proj_w = (const float*)d_in[1];
  const float* conv_w    = (const float*)d_in[2];
  const float* conv_b    = (const float*)d_in[3];
  const float* dt_bias   = (const float*)d_in[4];
  const float* A_log     = (const float*)d_in[5];
  const float* Dv        = (const float*)d_in[6];
  const float* norm_w    = (const float*)d_in[7];
  const float* out_proj_w= (const float*)d_in[8];
  float* out = (float*)d_out;

  char* base = (char*)d_ws;
  size_t o = 0;
  auto carve = [&](size_t bytes) { void* p = base + o; o = (o + bytes + 255) & ~(size_t)255; return p; };
  __bf16* hsb    = (__bf16*)carve((size_t)BL * D_MODEL * 2);
  __bf16* wib    = (__bf16*)carve((size_t)NPAD1 * D_MODEL * 2);
  __bf16* wob    = (__bf16*)carve((size_t)D_MODEL * D_INNER * 2);
  __bf16* zx     = (__bf16*)carve((size_t)BL * D_IN_PROJ * 2);
  __bf16* xbc    = (__bf16*)carve((size_t)BL * CONV_DIM * 2);
  __bf16* ynb    = (__bf16*)carve((size_t)BL * D_INNER * 2);
  float*  dtsp   = (float*)carve((size_t)BL * NHEADS * 4);
  float*  ybuf   = (float*)carve((size_t)BL * D_INNER * 4);
  float*  states = (float*)carve((size_t)B_SZ * NC * NHEADS * 128 * 128 * 4);
  float*  prevb  = (float*)carve((size_t)B_SZ * NC * NHEADS * 128 * 128 * 4);
  float*  cdec   = (float*)carve((size_t)B_SZ * NHEADS * NC * 4);

  to_bf16<<<(BL * D_MODEL) / 256, 256, 0, stream>>>(hs, hsb);
  w_pad_bf16<<<(NPAD1 * D_MODEL) / 256, 256, 0, stream>>>(in_proj_w, wib, D_IN_PROJ, D_MODEL);
  to_bf16<<<(D_MODEL * D_INNER) / 256, 256, 0, stream>>>(out_proj_w, wob);

  gemm_bf16_nt<__bf16><<<dim3(BL / 128, NPAD1 / 128), 256, 0, stream>>>(
      hsb, wib, zx, BL, D_IN_PROJ, D_MODEL);
  conv_silu<<<(BL * CONV_DIM) / 256, 256, 0, stream>>>(zx, conv_w, conv_b, xbc);
  dt_softplus<<<(BL * NHEADS) / 256, 256, 0, stream>>>(zx, dt_bias, dtsp);
  ssd_diag<<<B_SZ * NC * NHEADS * 4, 256, 0, stream>>>(xbc, dtsp, A_log, ybuf);
  ssd_states<<<B_SZ * NC * NHEADS, 256, 0, stream>>>(xbc, dtsp, A_log, states, cdec);
  ssd_scan<<<B_SZ * NHEADS, 256, 0, stream>>>(states, cdec, prevb);
  ssd_off<<<B_SZ * NC * NHEADS * 4, 256, 0, stream>>>(xbc, dtsp, A_log, prevb, Dv, ybuf);
  gatenorm<<<BL, 256, 0, stream>>>(ybuf, zx, norm_w, ynb);
  gemm_bf16_nt<float><<<dim3(BL / 128, D_MODEL / 128), 256, 0, stream>>>(
      ynb, wob, out, BL, D_MODEL, D_INNER);
}
